// RegionProposalNetwork_56590489092315
// MI455X (gfx1250) — matrix operand order repaired and checked by reading, not compiled, not run on hardware
//
#include <hip/hip_runtime.h>
#include <hip/hip_bf16.h>
#include <math.h>

// ---------------- CDNA5 WMMA types ----------------
typedef __attribute__((ext_vector_type(16))) _Float16 v16h;
typedef __attribute__((ext_vector_type(8)))  float    v8f;

#define BATCH   16
#define CIN     512
#define HW      2500      // 50*50
#define NANCH   22500     // 9*2500
#define MROWS   45        // 9 conf + 36 box channels
#define NTILE   157       // ceil(2500/16)
#define PRE_K   6000
#define POST_K  300
#define IMGSZ   800.0f

// =====================================================================
// Kernel 1: fused 1x1-conv heads as WMMA GEMM.
// One wave (32 threads) owns one 16-column block (N-tile) and ALL THREE
// 16-row M-tiles of D = W[45,512] x X[512,2500]: the B-fragment (features)
// is loaded once per K-step and fed to 3 chained v_wmma_f32_16x16x32_f16
// with independent accumulators. Feature traffic = minimal 82 MB.
// =====================================================================
__device__ __forceinline__ void store_tile(const v8f& acc, int mt, int b, int p, int half,
                                           const float* __restrict__ confB,
                                           const float* __restrict__ boxB,
                                           float* __restrict__ scoresRaw,
                                           float* __restrict__ bboxRaw)
{
    // C/D layout: lane = N (mod 16); VGPR r holds row M = r + 8*(lane/16).
    #pragma unroll
    for (int r = 0; r < 8; ++r) {
        int mm = mt * 16 + r + 8 * half;
        if (mm < MROWS) {
            float v = acc[r];
            if (mm < 9) {
                v += confB[mm];
                float s = 1.0f / (1.0f + __expf(-v));   // sigmoid
                scoresRaw[(size_t)b * NANCH + (size_t)mm * HW + p] = s;
            } else {
                int ch = mm - 9;
                bboxRaw[(size_t)b * (36*HW) + (size_t)ch * HW + p] = v + boxB[ch];
            }
        }
    }
}

__global__ __launch_bounds__(32)
void rpn_head_gemm(const float* __restrict__ feat,
                   const float* __restrict__ confW, const float* __restrict__ confB,
                   const float* __restrict__ boxW,  const float* __restrict__ boxB,
                   float* __restrict__ scoresRaw,   // [B][22500] sigmoid(conf)
                   float* __restrict__ bboxRaw)     // [B][36][2500]
{
    const int nt = blockIdx.x % NTILE;
    const int b  = blockIdx.x / NTILE;

    const int lane = threadIdx.x;     // 0..31 (wave32)
    const int half = lane >> 4;       // 0 or 1
    const int l16  = lane & 15;

    const int p  = nt * 16 + l16;         // output pixel (column)
    const int pc = (p < HW) ? p : (HW-1); // clamp OOB loads

    const float* fb = feat + (size_t)b * CIN * HW;

    // Weight row pointers for the three M-tiles (rows clamped to 44).
    const float* wrow[3];
    #pragma unroll
    for (int mt = 0; mt < 3; ++mt) {
        int m  = mt * 16 + l16;
        int mc = (m < MROWS) ? m : (MROWS - 1);
        wrow[mt] = (mc < 9) ? (confW + (size_t)mc * CIN)
                            : (boxW  + (size_t)(mc - 9) * CIN);
    }

    v8f acc0 = {}, acc1 = {}, acc2 = {};

    #pragma unroll 2
    for (int k0 = 0; k0 < CIN; k0 += 32) {
        // Prefetch next K-block's feature row for this lane (global_prefetch_b8).
        if (k0 + 32 < CIN)
            __builtin_prefetch(&fb[(size_t)(k0 + 32 + 16*half) * HW + pc], 0, 3);

        // B-fragment: lane = N (mod 16); elements 0-7 hold K=8*half+e, 8-15 hold K=16+8*half+(e-8).
        v16h bf;
        #pragma unroll
        for (int e = 0; e < 16; ++e) {
            int kk = (e < 8) ? (8*half + e) : (16 + 8*half + (e - 8));
            bf[e] = (_Float16)fb[(size_t)(k0 + kk) * HW + pc];
        }

        // A-fragments: 16-bit A 16x32 layout. lane = M (mod 16), half selects K groups.
        v16h a0, a1, a2;
        #pragma unroll
        for (int e = 0; e < 16; ++e) {
            int v = e >> 1, odd = e & 1;
            int kk = (v < 4) ? (8*half + 2*v + odd)
                             : (16 + 8*half + 2*(v-4) + odd);
            a0[e] = (_Float16)wrow[0][k0 + kk];
            a1[e] = (_Float16)wrow[1][k0 + kk];
            a2[e] = (_Float16)wrow[2][k0 + kk];
        }

        acc0 = __builtin_amdgcn_wmma_f32_16x16x32_f16(false, a0, false, bf,
                                                      (short)0, acc0, false, false);
        acc1 = __builtin_amdgcn_wmma_f32_16x16x32_f16(false, a1, false, bf,
                                                      (short)0, acc1, false, false);
        acc2 = __builtin_amdgcn_wmma_f32_16x16x32_f16(false, a2, false, bf,
                                                      (short)0, acc2, false, false);
    }

    if (p < HW) {
        store_tile(acc0, 0, b, p, half, confB, boxB, scoresRaw, bboxRaw);
        store_tile(acc1, 1, b, p, half, confB, boxB, scoresRaw, bboxRaw);
        store_tile(acc2, 2, b, p, half, confB, boxB, scoresRaw, bboxRaw);
    }
}

// =====================================================================
// Kernel 2: decode anchors + clip + min-size keep.
// reshape(B,36,2500)->(B,22500,4) is flat: offsets of anchor i are bboxRaw[b][4i..4i+3].
// keep encoded in the SIGN of the stored score (scores are in (0,1)):
// ranking uses fabs (pre-mask semantics), NMS restores -inf for negatives.
// =====================================================================
__global__ void rpn_decode(const float* __restrict__ bboxRaw,
                           float* __restrict__ scoresRW,    // in: sigmoid, out: signed-keep
                           float* __restrict__ boxes)       // [B][22500][4] clipped
{
    int idx = blockIdx.x * blockDim.x + threadIdx.x;
    if (idx >= BATCH * NANCH) return;
    int b = idx / NANCH, i = idx % NANCH;
    int a = i / HW, rem = i % HW;
    int y = rem / 50, x = rem % 50;
    int si = a / 3, ri = a % 3;

    const float scl[3] = {8.f, 16.f, 32.f};
    const float sqr[3] = {0.70710678118f, 1.0f, 1.41421356237f}; // sqrt(ratio)
    float size = scl[si] * 16.0f;
    float aw = size / sqr[ri];
    float ah = size * sqr[ri];
    float acx = (x + 0.5f) * 16.0f;
    float acy = (y + 0.5f) * 16.0f;

    const float4 o = ((const float4*)(bboxRaw + (size_t)b * (36*HW)))[i];
    float cx = acx + o.x * aw;
    float cy = acy + o.y * ah;
    float w  = __expf(o.z) * aw;
    float h  = __expf(o.w) * ah;
    float x1 = fminf(fmaxf(cx - 0.5f*w, 0.f), IMGSZ);
    float y1 = fminf(fmaxf(cy - 0.5f*h, 0.f), IMGSZ);
    float x2 = fminf(fmaxf(cx + 0.5f*w, 0.f), IMGSZ);
    float y2 = fminf(fmaxf(cy + 0.5f*h, 0.f), IMGSZ);
    bool keep = (x2 - x1 >= 0.001f) && (y2 - y1 >= 0.001f);

    float4 bx = {x1, y1, x2, y2};
    ((float4*)boxes)[(size_t)b * NANCH + i] = bx;
    float s = scoresRW[idx];
    scoresRW[idx] = keep ? s : -s;
}

// =====================================================================
// Kernel 3: per-batch top-6000 (bisection threshold) + ordered compaction
// + 300-step greedy NMS. One 1024-thread workgroup per batch.
// Candidate scores live in LDS; suppression = overwrite with -inf.
// =====================================================================
__global__ __launch_bounds__(1024)
void rpn_topk_nms(const float* __restrict__ scores,  // signed-keep
                  const float* __restrict__ boxes,
                  float* __restrict__ cboxG,         // [B][6000][4] scratch
                  float* __restrict__ out)           // kb | ks | ok
{
    __shared__ float sc[PRE_K];     // 24 KB candidate scores
    __shared__ float redS[1024];
    __shared__ int   redI[1024];
    __shared__ int   icnt;

    const int b = blockIdx.x;
    const int t = threadIdx.x;
    const float* S  = scores + (size_t)b * NANCH;
    const float* BX = boxes  + (size_t)b * NANCH * 4;
    float*       CB = cboxG  + (size_t)b * PRE_K * 4;

    // ---- Phase A: bisect threshold so count(|s| > thr) <= 6000 ----
    float lo = 0.f, hi = 1.f;
    for (int it = 0; it < 32; ++it) {
        float mid = 0.5f * (lo + hi);
        int c = 0;
        for (int i = t; i < NANCH; i += 1024) c += (fabsf(S[i]) > mid) ? 1 : 0;
        redI[t] = c;
        __syncthreads();
        for (int ofs = 512; ofs > 0; ofs >>= 1) {
            if (t < ofs) redI[t] += redI[t + ofs];
            __syncthreads();
        }
        int total = redI[0];
        __syncthreads();
        if (total > PRE_K) lo = mid; else hi = mid;
    }
    const float thr = hi;

    // ---- Phase B: ordered compaction (chunked Hillis-Steele scan) ----
    if (t == 0) icnt = 0;
    __syncthreads();
    for (int c0 = 0; c0 < NANCH; c0 += 1024) {
        int i = c0 + t;
        int flag = (i < NANCH && fabsf(S[i]) > thr) ? 1 : 0;
        redI[t] = flag;
        __syncthreads();
        for (int ofs = 1; ofs < 1024; ofs <<= 1) {
            int v = redI[t];
            int add = (t >= ofs) ? redI[t - ofs] : 0;
            __syncthreads();
            redI[t] = v + add;
            __syncthreads();
        }
        int incl = redI[t];
        int chunkTot = redI[1023];
        int pos = icnt + incl - flag;      // exclusive rank
        if (flag && pos < PRE_K) {
            float s = S[i];
            sc[pos] = (s > 0.f) ? s : -__builtin_inff();   // restore keep-mask
            ((float4*)CB)[pos] = ((const float4*)BX)[i];
        }
        __syncthreads();
        if (t == 0) icnt += chunkTot;
        __syncthreads();
    }
    const int cnt = (icnt < PRE_K) ? icnt : PRE_K;
    for (int i = cnt + t; i < PRE_K; i += 1024) sc[i] = -__builtin_inff();
    __syncthreads();

    // ---- Phase C: greedy NMS, 300 sequential selections ----
    float* outKB = out + (size_t)b * POST_K * 4;
    float* outKS = out + (size_t)BATCH * POST_K * 4 + (size_t)b * POST_K;
    float* outOK = out + (size_t)BATCH * POST_K * 5 + (size_t)b * POST_K;

    for (int iter = 0; iter < POST_K; ++iter) {
        // argmax (lowest index wins ties, matching jnp.argmax)
        float bs = -__builtin_inff(); int bi = 0x7fffffff;
        for (int i = t; i < cnt; i += 1024) {
            float s = sc[i];
            if (s > bs || (s == bs && i < bi)) { bs = s; bi = i; }
        }
        redS[t] = bs; redI[t] = bi;
        __syncthreads();
        for (int ofs = 512; ofs > 0; ofs >>= 1) {
            if (t < ofs) {
                float s2 = redS[t + ofs]; int i2 = redI[t + ofs];
                if (s2 > redS[t] || (s2 == redS[t] && i2 < redI[t])) {
                    redS[t] = s2; redI[t] = i2;
                }
            }
            __syncthreads();
        }
        float best = redS[0]; int bidx = redI[0];
        __syncthreads();

        bool ok = (best > -__builtin_inff()) && (bidx < cnt);
        float4 bb = {0.f, 0.f, 0.f, 0.f};
        if (ok) bb = ((const float4*)CB)[bidx];   // broadcast load (L2-resident)

        if (t == 0) {
            outKB[iter*4 + 0] = ok ? bb.x : 0.f;
            outKB[iter*4 + 1] = ok ? bb.y : 0.f;
            outKB[iter*4 + 2] = ok ? bb.z : 0.f;
            outKB[iter*4 + 3] = ok ? bb.w : 0.f;
            outKS[iter] = ok ? best : 0.f;
            outOK[iter] = ok ? 1.f  : 0.f;
            if (ok) sc[bidx] = -__builtin_inff();  // suppress selected
        }
        __syncthreads();

        if (ok) {
            float barea = (bb.z - bb.x) * (bb.w - bb.y);
            for (int i = t; i < cnt; i += 1024) {
                float s = sc[i];
                if (s == -__builtin_inff()) continue;
                float4 q = ((const float4*)CB)[i];
                float iw = fmaxf(fminf(q.z, bb.z) - fmaxf(q.x, bb.x), 0.f);
                float ih = fmaxf(fminf(q.w, bb.w) - fmaxf(q.y, bb.y), 0.f);
                float inter = iw * ih;
                float area  = (q.z - q.x) * (q.w - q.y);
                float uni   = area + barea - inter;
                float iou   = (uni > 0.f) ? (inter / uni) : 0.f;
                if (iou > 0.7f) sc[i] = -__builtin_inff();
            }
        }
        __syncthreads();
    }
}

// =====================================================================
extern "C" void kernel_launch(void* const* d_in, const int* in_sizes, int n_in,
                              void* d_out, int out_size, void* d_ws, size_t ws_size,
                              hipStream_t stream)
{
    (void)in_sizes; (void)n_in; (void)out_size; (void)ws_size;
    const float* feat  = (const float*)d_in[0];
    // d_in[1] images: only shape (800x800) used -> constant. d_in[2..3]: conv1 is dead.
    const float* confW = (const float*)d_in[4];
    const float* confB = (const float*)d_in[5];
    const float* boxW  = (const float*)d_in[6];
    const float* boxB  = (const float*)d_in[7];

    float* ws        = (float*)d_ws;
    float* scoresRaw = ws;                         //   360,000 f
    float* bboxRaw   = ws + 360000;                // 1,440,000 f
    float* boxes     = ws + 1800000;               // 1,440,000 f
    float* cbox      = ws + 3240000;               //   384,000 f  (total ~14.5 MB)
    float* out       = (float*)d_out;

    rpn_head_gemm<<<BATCH * NTILE, 32, 0, stream>>>(
        feat, confW, confB, boxW, boxB, scoresRaw, bboxRaw);

    int tot = BATCH * NANCH;
    rpn_decode<<<(tot + 255) / 256, 256, 0, stream>>>(bboxRaw, scoresRaw, boxes);

    rpn_topk_nms<<<BATCH, 1024, 0, stream>>>(scoresRaw, boxes, cbox, out);
}
